// DeepFusionBlock_31834297598796
// MI455X (gfx1250) — matrix operand order repaired and checked by reading, not compiled, not run on hardware
//
#include <hip/hip_runtime.h>

#define DEV __device__ __forceinline__

typedef __attribute__((ext_vector_type(16))) __bf16 v16bf;
typedef __attribute__((ext_vector_type(8)))  float  v8f;

constexpr int KNN = 16;
constexpr int CL  = 128;   // lidar channels
constexpr int CI  = 256;   // image channels
constexpr int H   = 128;   // hidden
constexpr int RT  = 64;    // row tile per block (4 WMMA M-tiles)

// ---- workspace layout (bytes), N = 65536 max ----
constexpr size_t NMAX   = 65536;
constexpr size_t OFF_QB = 0;                                // N*H bf16
constexpr size_t OFF_KB = OFF_QB + NMAX * H * 2;            // N*H bf16
constexpr size_t OFF_VB = OFF_KB + NMAX * H * 2;            // N*H bf16
constexpr size_t OFF_OH = OFF_VB + NMAX * H * 2;            // N*H bf16
constexpr size_t OFF_INV= OFF_OH + NMAX * H * 2;            // N int32
constexpr size_t OFF_WQT= OFF_INV + NMAX * 4;               // [H][CL] bf16
constexpr size_t OFF_WKT= OFF_WQT + (size_t)H * CL * 2;     // [H][CI] bf16
constexpr size_t OFF_WVT= OFF_WKT + (size_t)H * CI * 2;     // [H][CI] bf16
constexpr size_t OFF_WCT= OFF_WVT + (size_t)H * CI * 2;     // [CI][H] bf16

// ---- WMMA operand loaders (wave32, 16x16x32 bf16) ----
// A-matrix 16x32: lane l (m=l&15, h=l>>4): elems 0..7 = K[8h..8h+7],
//                 elems 8..15 = K[16+8h..16+8h+7]  -> two 16B chunks, gap 16 elems
DEV v16bf ldA(const __bf16* p) {
    union { v16bf v; uint4 u[2]; } t;
    t.u[0] = *reinterpret_cast<const uint4*>(p);
    t.u[1] = *reinterpret_cast<const uint4*>(p + 16);
    return t.v;
}
// B-matrix 32x16: lane l (n=l&15, h=l>>4): elems 0..7 = K[8h..8h+7],
//                 elems 8..15 = K[16+8h..16+8h+7]  -> two 16B chunks, gap 16 elems
DEV v16bf ldB(const __bf16* p) {
    union { v16bf v; uint4 u[2]; } t;
    t.u[0] = *reinterpret_cast<const uint4*>(p);
    t.u[1] = *reinterpret_cast<const uint4*>(p + 16);
    return t.v;
}
DEV v8f wmma_bf16(v16bf a, v16bf b, v8f c) {
    return __builtin_amdgcn_wmma_f32_16x16x32_bf16(
        false, a, false, b, (short)0, c, false, false);
}
// pack 4 f32 -> 4 bf16 (8 bytes)
DEV uint2 pack4(float4 f) {
    union { __bf16 b[4]; uint2 u; } t;
    t.b[0] = (__bf16)f.x; t.b[1] = (__bf16)f.y;
    t.b[2] = (__bf16)f.z; t.b[3] = (__bf16)f.w;
    return t.u;
}

// ---- kernel 1: convert weights to bf16, transposed so WMMA B reads rows ----
__global__ __launch_bounds__(256) void k_convert(
    const float* __restrict__ Wq, const float* __restrict__ Wk,
    const float* __restrict__ Wv, const float* __restrict__ Wc,
    char* __restrict__ ws) {
    __bf16* WqT = (__bf16*)(ws + OFF_WQT);   // [H][CL],  WqT[h][c] = Wq[c][h]
    __bf16* WkT = (__bf16*)(ws + OFF_WKT);   // [H][CI]
    __bf16* WvT = (__bf16*)(ws + OFF_WVT);   // [H][CI]
    __bf16* WcT = (__bf16*)(ws + OFF_WCT);   // [CI][H], WcT[n][h] = Wc[h][n]
    int i = blockIdx.x * 256 + threadIdx.x;
    if (i < H * CL) {
        int h = i / CL, c = i % CL;
        WqT[i] = (__bf16)Wq[c * H + h];
    }
    int j = i - H * CL;
    if (j >= 0 && j < H * CI) {
        int h = j / CI, c = j % CI;
        WkT[j] = (__bf16)Wk[c * H + h];
    }
    j -= H * CI;
    if (j >= 0 && j < H * CI) {
        int h = j / CI, c = j % CI;
        WvT[j] = (__bf16)Wv[c * H + h];
    }
    j -= H * CI;
    if (j >= 0 && j < CI * H) {
        int n = j / H, h = j % H;
        WcT[j] = (__bf16)Wc[h * CI + n];
    }
}

// ---- kernel 2: invalid-row flags (sum of image row == 0), wave per row ----
__global__ __launch_bounds__(256) void k_flags(
    const float* __restrict__ img, int* __restrict__ inv, int n) {
    int row  = (blockIdx.x * 256 + threadIdx.x) >> 5;
    int lane = threadIdx.x & 31;
    if (row >= n) return;
    const float* p = img + (size_t)row * CI;
    float s = 0.f;
    #pragma unroll
    for (int j = 0; j < CI / 32; ++j) s += p[lane + 32 * j];
    #pragma unroll
    for (int m = 16; m >= 1; m >>= 1) s += __shfl_xor(s, m, 32);
    if (lane == 0) inv[row] = (s == 0.0f) ? 1 : 0;
}

// ---- kernel 3: fused q/k/v projections via WMMA, 64-row tile per block ----
// 8 waves; wave w owns output columns [w*16, w*16+16); B-operands hoisted to
// registers and reused across the 4 M-tiles (4x less weight traffic).
__global__ __launch_bounds__(256) void k_qkv(
    const float* __restrict__ lidar, const float* __restrict__ image,
    const float* __restrict__ bq, const float* __restrict__ bk,
    const float* __restrict__ bv, char* __restrict__ ws) {
    __shared__ __bf16 sl[RT * CL];   // 16 KB, row stride CL
    __shared__ __bf16 si[RT * CI];   // 32 KB, row stride CI
    const int tid = threadIdx.x;
    const int m0  = blockIdx.x * RT;

    // tiles are contiguous row ranges -> flat vectorized fill with conversion
    {
        const float4* l4 = (const float4*)(lidar + (size_t)m0 * CL);
        const float4* i4 = (const float4*)(image + (size_t)m0 * CI);
        uint2* sl2 = (uint2*)sl;
        uint2* si2 = (uint2*)si;
        #pragma unroll
        for (int i = 0; i < RT * CL / 4 / 256; ++i)
            sl2[tid + 256 * i] = pack4(l4[tid + 256 * i]);
        #pragma unroll
        for (int i = 0; i < RT * CI / 4 / 256; ++i)
            si2[tid + 256 * i] = pack4(i4[tid + 256 * i]);
    }
    __syncthreads();

    const int w = tid >> 5, lane = tid & 31;
    const int nl = lane & 15, hh = lane >> 4;
    const int n0 = w * 16;                       // this wave's 16 output columns
    const __bf16* WqT = (const __bf16*)(ws + OFF_WQT);
    const __bf16* WkT = (const __bf16*)(ws + OFF_WKT);
    const __bf16* WvT = (const __bf16*)(ws + OFF_WVT);
    __bf16* qb = (__bf16*)(ws + OFF_QB);
    __bf16* kb = (__bf16*)(ws + OFF_KB);
    __bf16* vb = (__bf16*)(ws + OFF_VB);

    // --- q = lidar @ Wq ---
    {
        v16bf B[CL / 32];
        #pragma unroll
        for (int t = 0; t < CL / 32; ++t)
            B[t] = ldB(WqT + (size_t)(n0 + nl) * CL + t * 32 + 8 * hh);
        float bias = bq[n0 + nl];
        #pragma unroll
        for (int mt = 0; mt < RT / 16; ++mt) {
            v8f acc = {};
            #pragma unroll
            for (int t = 0; t < CL / 32; ++t)
                acc = wmma_bf16(ldA(sl + (mt * 16 + nl) * CL + t * 32 + 8 * hh),
                                B[t], acc);
            #pragma unroll
            for (int r = 0; r < 8; ++r)
                qb[(size_t)(m0 + mt * 16 + r + 8 * hh) * H + n0 + nl] =
                    (__bf16)(acc[r] + bias);
        }
    }
    // --- k = image @ Wk ---
    {
        v16bf B[CI / 32];
        #pragma unroll
        for (int t = 0; t < CI / 32; ++t)
            B[t] = ldB(WkT + (size_t)(n0 + nl) * CI + t * 32 + 8 * hh);
        float bias = bk[n0 + nl];
        #pragma unroll
        for (int mt = 0; mt < RT / 16; ++mt) {
            v8f acc = {};
            #pragma unroll
            for (int t = 0; t < CI / 32; ++t)
                acc = wmma_bf16(ldA(si + (mt * 16 + nl) * CI + t * 32 + 8 * hh),
                                B[t], acc);
            #pragma unroll
            for (int r = 0; r < 8; ++r)
                kb[(size_t)(m0 + mt * 16 + r + 8 * hh) * H + n0 + nl] =
                    (__bf16)(acc[r] + bias);
        }
    }
    // --- v = image @ Wv ---
    {
        v16bf B[CI / 32];
        #pragma unroll
        for (int t = 0; t < CI / 32; ++t)
            B[t] = ldB(WvT + (size_t)(n0 + nl) * CI + t * 32 + 8 * hh);
        float bias = bv[n0 + nl];
        #pragma unroll
        for (int mt = 0; mt < RT / 16; ++mt) {
            v8f acc = {};
            #pragma unroll
            for (int t = 0; t < CI / 32; ++t)
                acc = wmma_bf16(ldA(si + (mt * 16 + nl) * CI + t * 32 + 8 * hh),
                                B[t], acc);
            #pragma unroll
            for (int r = 0; r < 8; ++r)
                vb[(size_t)(m0 + mt * 16 + r + 8 * hh) * H + n0 + nl] =
                    (__bf16)(acc[r] + bias);
        }
    }
}

// ---- kernel 4: KNN gather + masked softmax + weighted sum; wave per point ----
__global__ __launch_bounds__(256) void k_attn(
    const int* __restrict__ knn, char* __restrict__ ws, int n) {
    int pt   = (blockIdx.x * 256 + threadIdx.x) >> 5;
    int lane = threadIdx.x & 31;
    if (pt >= n) return;
    const __bf16* qb = (const __bf16*)(ws + OFF_QB);
    const __bf16* kb = (const __bf16*)(ws + OFF_KB);
    const __bf16* vb = (const __bf16*)(ws + OFF_VB);
    const int*    inv= (const int*)(ws + OFF_INV);
    __bf16*       oh = (__bf16*)(ws + OFF_OH);

    float q[4];
    const __bf16* qr = qb + (size_t)pt * H;
    #pragma unroll
    for (int j = 0; j < 4; ++j) q[j] = (float)qr[lane + 32 * j];

    const float scale = 0.08838834764831845f;   // 1/sqrt(128)
    float att[KNN]; int ids[KNN];
    #pragma unroll
    for (int kk = 0; kk < KNN; ++kk) {
        int id = knn[(size_t)pt * KNN + kk];
        ids[kk] = id;
        const __bf16* kr = kb + (size_t)id * H;
        __builtin_prefetch(vb + (size_t)id * H, 0, 0);   // warm v row (L2-resident)
        float s = 0.f;
        #pragma unroll
        for (int j = 0; j < 4; ++j) s += q[j] * (float)kr[lane + 32 * j];
        #pragma unroll
        for (int m = 16; m >= 1; m >>= 1) s += __shfl_xor(s, m, 32);
        att[kk] = inv[id] ? -__builtin_inff() : s * scale;
    }
    float mx = -__builtin_inff();
    #pragma unroll
    for (int kk = 0; kk < KNN; ++kk) mx = fmaxf(mx, att[kk]);
    float o[4] = {0.f, 0.f, 0.f, 0.f};
    if (mx != -__builtin_inff()) {              // wave-uniform branch
        float ssum = 0.f;
        #pragma unroll
        for (int kk = 0; kk < KNN; ++kk) { att[kk] = __expf(att[kk] - mx); ssum += att[kk]; }
        float rs = 1.0f / ssum;
        #pragma unroll
        for (int kk = 0; kk < KNN; ++kk) {
            float wgt = att[kk] * rs;
            const __bf16* vr = vb + (size_t)ids[kk] * H;
            #pragma unroll
            for (int j = 0; j < 4; ++j) o[j] += wgt * (float)vr[lane + 32 * j];
        }
    }
    __bf16* orow = oh + (size_t)pt * H;
    #pragma unroll
    for (int j = 0; j < 4; ++j) orow[lane + 32 * j] = (__bf16)o[j];
}

// ---- kernel 5: out = out_h @ Wc + bc  (N x 128 @ 128 x 256 -> f32) ----
// 64-row tile per block; 8 waves x 2 column tiles = 256 output columns.
__global__ __launch_bounds__(256) void k_out(
    const float* __restrict__ bc, float* __restrict__ out, char* __restrict__ ws) {
    __shared__ __bf16 sa[RT * H];    // 16 KB
    const int tid = threadIdx.x;
    const int m0  = blockIdx.x * RT;
    const __bf16* oh  = (const __bf16*)(ws + OFF_OH);
    const __bf16* WcT = (const __bf16*)(ws + OFF_WCT);

    {   // contiguous flat copy, 16B vectors
        const uint4* src = (const uint4*)(oh + (size_t)m0 * H);
        uint4* dst = (uint4*)sa;
        #pragma unroll
        for (int i = 0; i < RT * H / 8 / 256; ++i)
            dst[tid + 256 * i] = src[tid + 256 * i];
    }
    __syncthreads();

    const int w = tid >> 5, lane = tid & 31;
    const int nl = lane & 15, hh = lane >> 4;
    #pragma unroll
    for (int ct = 0; ct < 2; ++ct) {
        const int n0 = w * 16 + ct * 128;
        v16bf B[H / 32];
        #pragma unroll
        for (int t = 0; t < H / 32; ++t)
            B[t] = ldB(WcT + (size_t)(n0 + nl) * H + t * 32 + 8 * hh);
        float bias = bc[n0 + nl];
        #pragma unroll
        for (int mt = 0; mt < RT / 16; ++mt) {
            v8f acc = {};
            #pragma unroll
            for (int t = 0; t < H / 32; ++t)
                acc = wmma_bf16(ldA(sa + (mt * 16 + nl) * H + t * 32 + 8 * hh),
                                B[t], acc);
            #pragma unroll
            for (int r = 0; r < 8; ++r)
                out[(size_t)(m0 + mt * 16 + r + 8 * hh) * CI + n0 + nl] =
                    acc[r] + bias;
        }
    }
}

extern "C" void kernel_launch(void* const* d_in, const int* in_sizes, int n_in,
                              void* d_out, int out_size, void* d_ws, size_t ws_size,
                              hipStream_t stream) {
    const float* lidar = (const float*)d_in[0];
    const float* image = (const float*)d_in[1];
    const float* Wq    = (const float*)d_in[2];
    const float* bq    = (const float*)d_in[3];
    const float* Wk    = (const float*)d_in[4];
    const float* bk    = (const float*)d_in[5];
    const float* Wv    = (const float*)d_in[6];
    const float* bv    = (const float*)d_in[7];
    const float* Wc    = (const float*)d_in[8];
    const float* bc    = (const float*)d_in[9];
    const int*   knn   = (const int*)d_in[10];
    float* out = (float*)d_out;
    char*  ws  = (char*)d_ws;

    const int n = in_sizes[0] / CL;              // 65536

    const int conv_elems = H * CL + 2 * H * CI + CI * H;   // 114688
    k_convert<<<(conv_elems + 255) / 256, 256, 0, stream>>>(Wq, Wk, Wv, Wc, ws);
    k_flags<<<(n * 32 + 255) / 256, 256, 0, stream>>>(image, (int*)(ws + OFF_INV), n);
    k_qkv<<<n / RT, 256, 0, stream>>>(lidar, image, bq, bk, bv, ws);
    k_attn<<<(n * 32 + 255) / 256, 256, 0, stream>>>(knn, ws, n);
    k_out<<<n / RT, 256, 0, stream>>>(bc, out, ws);
}
